// MultiHeadAttention_42176578847463
// MI455X (gfx1250) — compile-verified, matrix operand order checked by reading, not run on hardware
//
#include <hip/hip_runtime.h>
#include <cstdint>
#include <cstddef>

// MI455X / gfx1250, wave32. All GEMM-shaped math via v_wmma_f32_16x16x32_bf16.
// GEMM waves own 64x64 tiles (4x4 accumulators): 8 fragment loads per 16 WMMAs.

#define D_MODEL 1024
#define NHEAD   16
#define HD      64
#define BATCH   2
#define SEQ     2048
#define MROWS   (BATCH * SEQ)      // 4096
#define N_QKV   (3 * D_MODEL)      // 3072

typedef __attribute__((ext_vector_type(16))) __bf16 v16bf;
typedef __attribute__((ext_vector_type(8)))  float  v8f;

union Frag16 { v16bf v; uint4 q[2]; };

__device__ __forceinline__ uint16_t f32_to_bf16(float f) {
    union { float f; uint32_t u; } c; c.f = f;
    uint32_t u = c.u;
    return (uint16_t)((u + 0x7FFFu + ((u >> 16) & 1u)) >> 16);
}

__device__ __forceinline__ v8f zero_v8f() {
    v8f z = {0.f, 0.f, 0.f, 0.f, 0.f, 0.f, 0.f, 0.f};
    return z;
}

__device__ __forceinline__ v8f wmma_bf16(v16bf a, v16bf b, v8f c) {
    // (neg_a, A, neg_b, B, c_mod, C, reuse_a, reuse_b)
    return __builtin_amdgcn_wmma_f32_16x16x32_bf16(false, a, false, b,
                                                   (short)0, c, false, false);
}

// A/B fragment for 16x16x32 bf16 WMMA from a row-major [rows x ld] bf16 array.
// lane<16 : row=row0+lane,     K = k0 + {0..7, 16..23}
// lane>=16: row=row0+lane-16,  K = k0 + {8..15, 24..31}
// -> two 16-byte contiguous loads per lane (global_load_b128 / ds_load_b128).
__device__ __forceinline__ v16bf load_frag(const uint16_t* base, int row0, int k0,
                                           int ld, int lane) {
    const int half = lane >> 4;
    const uint16_t* p = base + (size_t)(row0 + (lane & 15)) * ld + k0 + half * 8;
    Frag16 f;
    f.q[0] = *(const uint4*)(p);
    f.q[1] = *(const uint4*)(p + 16);
    return f.v;
}

// 64x64 output tile per wave: acc[4][4], K-loop over `K` with 32-deep WMMA.
__device__ __forceinline__ void gemm_tile_64x64(const uint16_t* __restrict__ A,
                                                const uint16_t* __restrict__ B,
                                                int m0, int n0, int K, int lane,
                                                v8f acc[4][4]) {
#pragma unroll
    for (int i = 0; i < 4; i++)
#pragma unroll
        for (int j = 0; j < 4; j++) acc[i][j] = zero_v8f();

    for (int k0 = 0; k0 < K; k0 += 32) {
        if (k0 + 32 < K) {   // global_prefetch_b8 next K-slab
            __builtin_prefetch(A + (size_t)(m0 + (lane & 15)) * K + k0 + 32, 0, 1);
            __builtin_prefetch(B + (size_t)(n0 + (lane & 15)) * K + k0 + 32, 0, 1);
        }
        v16bf a[4], b[4];
#pragma unroll
        for (int i = 0; i < 4; i++) a[i] = load_frag(A, m0 + i * 16, k0, K, lane);
#pragma unroll
        for (int j = 0; j < 4; j++) b[j] = load_frag(B, n0 + j * 16, k0, K, lane);
#pragma unroll
        for (int i = 0; i < 4; i++)
#pragma unroll
            for (int j = 0; j < 4; j++)
                acc[i][j] = wmma_bf16(a[i], b[j], acc[i][j]);
    }
}

// ---------------------------------------------------------------- cast kernels
__global__ void cast_f32_bf16_kernel(const float* __restrict__ in,
                                     uint16_t* __restrict__ out, int n) {
    int i = blockIdx.x * blockDim.x + threadIdx.x;
    if (i < n) out[i] = f32_to_bf16(in[i]);
}

// W is [K x N] row-major fp32; write W^T as [N x K] row-major bf16.
__global__ void transpose_cast_kernel(const float* __restrict__ w,
                                      uint16_t* __restrict__ wt, int K, int N) {
    int i = blockIdx.x * blockDim.x + threadIdx.x;
    if (i < K * N) {
        int k = i / N, n = i - k * N;
        wt[(size_t)n * K + k] = f32_to_bf16(w[i]);
    }
}

// ------------------------------------------------------------- QKV projection
// C[4096 x 3072] = xb[4096 x 1024] @ Wqkv + b ; scatter into q / k / v^T (bf16)
// 8 waves: 2 in M x 4 in N -> workgroup tile 128 x 256.
__global__ void __launch_bounds__(256)
qkv_gemm_kernel(const uint16_t* __restrict__ xb, const uint16_t* __restrict__ wt,
                const float* __restrict__ bias,
                uint16_t* __restrict__ qo, uint16_t* __restrict__ ko,
                uint16_t* __restrict__ vto) {
    const int lane = threadIdx.x & 31;
    const int wid  = threadIdx.x >> 5;
    const int m0 = blockIdx.y * 128 + (wid & 1) * 64;
    const int n0 = blockIdx.x * 256 + (wid >> 1) * 64;

    v8f acc[4][4];
    gemm_tile_64x64(xb, wt, m0, n0, D_MODEL, lane, acc);

    // C/D layout: element (m,n): lane = (n&15) + (m>=8)*16, vgpr = m&7
    const int nl = lane & 15;
    const int mh = (lane >> 4) * 8;
#pragma unroll
    for (int i = 0; i < 4; i++) {
#pragma unroll
        for (int j = 0; j < 4; j++) {
            const int n   = n0 + j * 16 + nl;
            const int seg = n >> 10;         // 0=q 1=k 2=v
            const int c   = n & (D_MODEL - 1);
            const int h   = c >> 6;
            const int d   = c & (HD - 1);
            const float bv = bias[n];
#pragma unroll
            for (int r = 0; r < 8; r++) {
                const int m  = m0 + i * 16 + mh + r;
                const int bb = m >> 11;      // batch (T = 2048)
                const int t  = m & (SEQ - 1);
                const int bh = bb * NHEAD + h;
                float v = acc[i][j][r] + bv;
                if (seg == 0)
                    qo[((size_t)bh * SEQ + t) * HD + d] = f32_to_bf16(v * 0.125f); // fold 1/sqrt(64)
                else if (seg == 1)
                    ko[((size_t)bh * SEQ + t) * HD + d] = f32_to_bf16(v);
                else
                    vto[((size_t)bh * HD + d) * SEQ + t] = f32_to_bf16(v);         // V transposed
            }
        }
    }
}

// ------------------------------------------------- flash-style causal attention
// One wave = 16 query rows; online softmax; P@V via LDS round-trip for layout.
__global__ void __launch_bounds__(128)
attn_kernel(const uint16_t* __restrict__ q, const uint16_t* __restrict__ k,
            const uint16_t* __restrict__ vt, uint16_t* __restrict__ y) {
    __shared__ __align__(16) uint16_t ldsp[4][16][32];

    const int lane = threadIdx.x & 31;
    const int wid  = threadIdx.x >> 5;
    const int bh   = blockIdx.y;
    const int bat  = bh >> 4;
    const int h    = bh & (NHEAD - 1);
    const int q0   = blockIdx.x * 64 + wid * 16;

    const uint16_t* qp = q  + (size_t)bh * SEQ * HD;
    const uint16_t* kp = k  + (size_t)bh * SEQ * HD;
    const uint16_t* vp = vt + (size_t)bh * HD * SEQ;

    const int nl = lane & 15;
    const int mh = (lane >> 4) * 8;

    v8f O[4];
#pragma unroll
    for (int j = 0; j < 4; j++) O[j] = zero_v8f();
    float rm[8], rs[8];
#pragma unroll
    for (int r = 0; r < 8; r++) { rm[r] = -3.0e38f; rs[r] = 0.f; }

    // Q fragment covers all of hd=64 (two K-slabs); scale already folded in.
    v16bf aq0 = load_frag(qp, q0, 0,  HD, lane);
    v16bf aq1 = load_frag(qp, q0, 32, HD, lane);

    for (int c0 = 0; c0 <= q0 + 15; c0 += 32) {
        // S tile: 16 queries x 32 keys = two 16x16 accumulators
        v8f s[2] = { zero_v8f(), zero_v8f() };
#pragma unroll
        for (int ss = 0; ss < 2; ss++) {
            v16bf bk0 = load_frag(kp, c0 + ss * 16, 0,  HD, lane);
            v16bf bk1 = load_frag(kp, c0 + ss * 16, 32, HD, lane);
            s[ss] = wmma_bf16(aq0, bk0, s[ss]);
            s[ss] = wmma_bf16(aq1, bk1, s[ss]);
        }

        float ml[8], al[8], psum[8];
#pragma unroll
        for (int r = 0; r < 8; r++) {
            const int qi = q0 + mh + r;
            if (c0 + nl      > qi) s[0][r] = -3.0e38f;   // causal mask
            if (c0 + 16 + nl > qi) s[1][r] = -3.0e38f;
            ml[r] = fmaxf(s[0][r], s[1][r]);
        }
        // row max: reduce across the 16 lanes holding one row
#pragma unroll
        for (int off = 1; off < 16; off <<= 1)
#pragma unroll
            for (int r = 0; r < 8; r++)
                ml[r] = fmaxf(ml[r], __shfl_xor(ml[r], off, 16));
#pragma unroll
        for (int r = 0; r < 8; r++) {
            const float mn = fmaxf(rm[r], ml[r]);
            al[r] = __expf(rm[r] - mn);
            rm[r] = mn;
            const float p0 = __expf(s[0][r] - mn);
            const float p1 = __expf(s[1][r] - mn);
            s[0][r] = p0; s[1][r] = p1;
            psum[r] = p0 + p1;
        }
#pragma unroll
        for (int off = 1; off < 16; off <<= 1)
#pragma unroll
            for (int r = 0; r < 8; r++)
                psum[r] += __shfl_xor(psum[r], off, 16);
#pragma unroll
        for (int r = 0; r < 8; r++) rs[r] = rs[r] * al[r] + psum[r];
#pragma unroll
        for (int j = 0; j < 4; j++)
#pragma unroll
            for (int r = 0; r < 8; r++) O[j][r] *= al[r];

        // P (16x32 bf16) -> LDS, reload in A-fragment layout (per-wave region,
        // intra-wave DS ordering; compiler inserts s_wait_dscnt)
#pragma unroll
        for (int ss = 0; ss < 2; ss++)
#pragma unroll
            for (int r = 0; r < 8; r++)
                ldsp[wid][mh + r][ss * 16 + nl] = f32_to_bf16(s[ss][r]);
        v16bf pa = load_frag(&ldsp[wid][0][0], 0, 0, 32, lane);

        // O += P @ V : B operand from v^T [hd x T], contiguous along keys
#pragma unroll
        for (int j = 0; j < 4; j++) {
            v16bf vb = load_frag(vp, j * 16, c0, SEQ, lane);
            O[j] = wmma_bf16(pa, vb, O[j]);
        }
    }

    // y[B,T,D] bf16: row = bat*SEQ + (q0+m), col = h*64 + j*16 + n
#pragma unroll
    for (int j = 0; j < 4; j++) {
        const int col = h * HD + j * 16 + nl;
#pragma unroll
        for (int r = 0; r < 8; r++) {
            const int qi = q0 + mh + r;
            y[((size_t)bat * SEQ + qi) * D_MODEL + col] = f32_to_bf16(O[j][r] / rs[r]);
        }
    }
}

// ------------------------------------------------------------ output projection
__global__ void __launch_bounds__(256)
out_gemm_kernel(const uint16_t* __restrict__ yb, const uint16_t* __restrict__ wt,
                const float* __restrict__ bias, float* __restrict__ out) {
    const int lane = threadIdx.x & 31;
    const int wid  = threadIdx.x >> 5;
    const int m0 = blockIdx.y * 128 + (wid & 1) * 64;
    const int n0 = blockIdx.x * 256 + (wid >> 1) * 64;

    v8f acc[4][4];
    gemm_tile_64x64(yb, wt, m0, n0, D_MODEL, lane, acc);

    const int nl = lane & 15;
    const int mh = (lane >> 4) * 8;
#pragma unroll
    for (int i = 0; i < 4; i++)
#pragma unroll
        for (int j = 0; j < 4; j++) {
            const int n = n0 + j * 16 + nl;
            const float bv = bias[n];
#pragma unroll
            for (int r = 0; r < 8; r++) {
                const int m = m0 + i * 16 + mh + r;
                out[(size_t)m * D_MODEL + n] = acc[i][j][r] + bv;
            }
        }
}

// ------------------------------------------------------------------- launcher
extern "C" void kernel_launch(void* const* d_in, const int* in_sizes, int n_in,
                              void* d_out, int out_size, void* d_ws, size_t ws_size,
                              hipStream_t stream) {
    const float* x    = (const float*)d_in[0];
    const float* Wqkv = (const float*)d_in[1];
    const float* bqkv = (const float*)d_in[2];
    const float* Wout = (const float*)d_in[3];
    const float* bout = (const float*)d_in[4];
    // d_in[5] = causal mask: handled analytically in attn_kernel
    float* out = (float*)d_out;

    uint8_t* ws = (uint8_t*)d_ws;
    uint16_t* xb  = (uint16_t*)ws; ws += (size_t)MROWS * D_MODEL * 2;     //  8 MiB
    uint16_t* wqt = (uint16_t*)ws; ws += (size_t)N_QKV * D_MODEL * 2;     //  6 MiB
    uint16_t* wot = (uint16_t*)ws; ws += (size_t)D_MODEL * D_MODEL * 2;   //  2 MiB
    uint16_t* qb  = (uint16_t*)ws; ws += (size_t)MROWS * D_MODEL * 2;     //  8 MiB
    uint16_t* kb  = (uint16_t*)ws; ws += (size_t)MROWS * D_MODEL * 2;     //  8 MiB
    uint16_t* vtb = (uint16_t*)ws; ws += (size_t)MROWS * D_MODEL * 2;     //  8 MiB
    uint16_t* yb  = (uint16_t*)ws; ws += (size_t)MROWS * D_MODEL * 2;     //  8 MiB

    cast_f32_bf16_kernel<<<(MROWS * D_MODEL) / 256, 256, 0, stream>>>(
        x, xb, MROWS * D_MODEL);
    transpose_cast_kernel<<<(D_MODEL * N_QKV) / 256, 256, 0, stream>>>(
        Wqkv, wqt, D_MODEL, N_QKV);
    transpose_cast_kernel<<<(D_MODEL * D_MODEL) / 256, 256, 0, stream>>>(
        Wout, wot, D_MODEL, D_MODEL);

    qkv_gemm_kernel<<<dim3(N_QKV / 256, MROWS / 128), 256, 0, stream>>>(
        xb, wqt, bqkv, qb, kb, vtb);

    attn_kernel<<<dim3(SEQ / 64, BATCH * NHEAD), 128, 0, stream>>>(
        qb, kb, vtb, yb);

    out_gemm_kernel<<<dim3(D_MODEL / 256, MROWS / 128), 256, 0, stream>>>(
        yb, wot, bout, out);
}
